// ConvTest_PCFNet_30537217475338
// MI455X (gfx1250) — matrix operand order repaired and checked by reading, not compiled, not run on hardware
//
#include <hip/hip_runtime.h>
#include <hip/hip_bf16.h>
#include <math.h>

// ---------------------------------------------------------------------------
// PCFNet forward on MI455X (gfx1250, wave32, WMMA 16x16x32 f16 -> f32).
//
// conv2 (30 GFLOP) and fc1 (3.3 GFLOP) run on v_wmma_f32_16x16x32_f16.
// Shared operands (conv2 weight fragments, fc1 activation rows) are staged
// into LDS with CDNA5 async global->LDS loads (ASYNCcnt) and read back with
// ds_load, so WMMAs are not serialized behind global load waits.
// conv2 fuses bias+relu+2x2 maxpool via __shfl_xor.
// ---------------------------------------------------------------------------

typedef __attribute__((ext_vector_type(16))) _Float16 v16h;
typedef __attribute__((ext_vector_type(8)))  _Float16 v8h;
typedef __attribute__((ext_vector_type(8)))  float    v8f;

#define BATCH 4096

__device__ __forceinline__ void async_copy_b128(unsigned lds_addr, unsigned long long gaddr) {
  asm volatile("global_load_async_to_lds_b128 %0, %1, off"
               :: "v"(lds_addr), "v"(gaddr) : "memory");
}
__device__ __forceinline__ void wait_async0() {
  asm volatile("s_wait_asynccnt 0x0" ::: "memory");
}

// ---- Gabor bank + identity maps: w1[32][3][3] f32 -------------------------
__global__ void k_gabor(const float* __restrict__ theta, const float* __restrict__ sigma,
                        const float* __restrict__ gamma, const float* __restrict__ lambd,
                        const float* __restrict__ psi, float* __restrict__ w1) {
  int t = threadIdx.x;
  if (t >= 32 * 9) return;
  int oc = t / 9, pos = t % 9, i = pos / 3, j = pos % 3;
  float val;
  if (oc < 16) {
    float th = theta[oc], sg = sigma[oc], gm = gamma[oc], lb = lambd[oc], ps = psi[oc];
    float xg = (float)(i - 1), yg = (float)(j - 1);
    float c = cosf(th), s = sinf(th);
    float xt = xg * c + yg * s;
    float yt = -xg * s + yg * c;
    float sx = sg, sy = sg / gm;
    float env = expf(-0.5f * (xt * xt / (sx * sx) + yt * yt / (sy * sy)));
    float car = cosf(2.0f * 3.14159265358979323846f * xt / lb + ps);
    val = env * car;
  } else {
    val = (i == 1 && j == 1) ? 1.0f : 0.0f;
  }
  w1[t] = val;
}

// ---- zero-fill (for padded NHWC buffer borders) ---------------------------
__global__ void k_zero(uint4* __restrict__ p, int n4) {
  int t = blockIdx.x * blockDim.x + threadIdx.x;
  if (t < n4) p[t] = make_uint4(0u, 0u, 0u, 0u);
}

// ---- repack conv2_w (64,32,3,3) into WMMA A-fragment lane layout ----------
// K ordering: k = kstep*32 + ic, kstep j <-> (kh,kw)=(j/3,j%3).
// A element (lane L, half h): M = L&15, klocal = h + (L>>4)*16.
__global__ void k_repack_conv2w(const float* __restrict__ w, _Float16* __restrict__ afrag) {
  int t = blockIdx.x * blockDim.x + threadIdx.x;
  if (t >= 4 * 9 * 32 * 16) return;
  int h = t & 15;
  int lane = (t >> 4) & 31;
  int j = (t >> 9) % 9;
  int mt = t / 4608;
  int part = lane >> 4;
  int m = lane & 15;
  int klocal = h + part * 16;
  int oc = mt * 16 + m, ic = klocal, kh = j / 3, kw = j % 3;
  afrag[t] = (_Float16)w[((oc * 32 + ic) * 3 + kh) * 3 + kw];
}

// ---- f32 -> f16 convert (fc1 weights) -------------------------------------
__global__ void k_f32_to_f16(const float* __restrict__ src, _Float16* __restrict__ dst, int n) {
  int t = blockIdx.x * blockDim.x + threadIdx.x;
  if (t < n) dst[t] = (_Float16)src[t];
}

// ---- conv1 + relu + 2x2 maxpool -> padded NHWC f16 ------------------------
__global__ void k_conv1(const float* __restrict__ x, const float* __restrict__ w1,
                        _Float16* __restrict__ xpad) {
  int t = blockIdx.x * blockDim.x + threadIdx.x;
  if (t >= BATCH * 14 * 14) return;
  int b = t / 196, rr = t % 196, y2 = rr / 14, x2 = rr % 14;

  float win[4][4];
  const float* xb = x + b * 784;
#pragma unroll
  for (int r = 0; r < 4; ++r) {
    int yy = 2 * y2 - 1 + r;
#pragma unroll
    for (int c = 0; c < 4; ++c) {
      int xx = 2 * x2 - 1 + c;
      win[r][c] = (yy >= 0 && yy < 28 && xx >= 0 && xx < 28) ? xb[yy * 28 + xx] : 0.0f;
    }
  }
  _Float16* outp = xpad + ((b * 16 + (y2 + 1)) * 16 + (x2 + 1)) * 32;
  for (int oc = 0; oc < 32; ++oc) {
    float w[9];
#pragma unroll
    for (int k = 0; k < 9; ++k) w[k] = w1[oc * 9 + k];
    float best = 0.0f;  // relu folded into pool-max
#pragma unroll
    for (int dy = 0; dy < 2; ++dy)
#pragma unroll
      for (int dx = 0; dx < 2; ++dx) {
        float s = 0.0f;
#pragma unroll
        for (int kh = 0; kh < 3; ++kh)
#pragma unroll
          for (int kw = 0; kw < 3; ++kw)
            s = fmaf(w[kh * 3 + kw], win[dy + kh][dx + kw], s);
        best = fmaxf(best, s);
      }
    outp[oc] = (_Float16)best;
  }
}

// ---- conv2 implicit GEMM via WMMA, fused bias+relu+maxpool ----------------
// M=64 (4 tiles), K=288 (9 ksteps x 32 ic), 2 N-tiles (2x16 positions) per
// wave. A fragments (36 KB, identical for all waves) staged in LDS via async
// global->LDS loads; each LDS A fragment feeds 2 independent WMMA chains.
__global__ void __launch_bounds__(256)
k_conv2_wmma(const _Float16* __restrict__ xpad, const _Float16* __restrict__ afrag,
             const float* __restrict__ bias, _Float16* __restrict__ act1) {
  __shared__ _Float16 sA[4 * 9 * 32 * 16];  // 36864 B

  // stage A fragments: 2304 x 16B chunks, 256 threads x 9 chunks
  {
    unsigned lds0 = (unsigned)(unsigned long long)&sA[0];
    unsigned long long g0 = (unsigned long long)afrag;
#pragma unroll
    for (int i = 0; i < 9; ++i) {
      unsigned chunk = threadIdx.x + i * 256;
      async_copy_b128(lds0 + chunk * 16u, g0 + (unsigned long long)chunk * 16ull);
    }
    wait_async0();
  }
  __syncthreads();

  int lane = threadIdx.x & 31;
  int wave = threadIdx.x >> 5;
  int tpair = blockIdx.x * 8 + wave;  // 0..25087
  int n = lane & 15;
  int part = lane >> 4;

  int base[2], outbase[2];
#pragma unroll
  for (int u = 0; u < 2; ++u) {
    int tile = tpair * 2 + u;          // 0..50175
    int cell = tile * 4 + (n >> 2);    // pooled cell
    int sub = n & 3, dy = sub >> 1, dx = sub & 1;
    int b = cell / 49, rc = cell % 49, y2 = rc / 7, x2 = rc % 7;
    int y = y2 * 2 + dy, xx = x2 * 2 + dx;  // conv position 0..13
    base[u] = ((b * 16 + y) * 16 + xx) * 32 + part * 16;
    outbase[u] = b * 3136 + y2 * 7 + x2;
  }

  v8f zero = {0.f, 0.f, 0.f, 0.f, 0.f, 0.f, 0.f, 0.f};
  v8f acc[2][4] = {{zero, zero, zero, zero}, {zero, zero, zero, zero}};

#pragma unroll
  for (int j = 0; j < 9; ++j) {
    int koff = ((j / 3) * 16 + (j % 3)) * 32;
    v16h b0 = *(const v16h*)(xpad + base[0] + koff);
    v16h b1 = *(const v16h*)(xpad + base[1] + koff);
#pragma unroll
    for (int m = 0; m < 4; ++m) {
      v16h am = *(const v16h*)(&sA[((m * 9 + j) * 32 + lane) * 16]);
      acc[0][m] = __builtin_amdgcn_wmma_f32_16x16x32_f16(
          false, am, false, b0, (short)0, acc[0][m], false, false);
      acc[1][m] = __builtin_amdgcn_wmma_f32_16x16x32_f16(
          false, am, false, b1, (short)0, acc[1][m], false, false);
    }
  }

#pragma unroll
  for (int u = 0; u < 2; ++u) {
#pragma unroll
    for (int m = 0; m < 4; ++m) {
#pragma unroll
      for (int r = 0; r < 8; ++r) {
        int oc = m * 16 + r + part * 8;
        float v = fmaxf(acc[u][m][r] + bias[oc], 0.0f);
        v = fmaxf(v, __shfl_xor(v, 1, 32));  // maxpool across 2x2 (4 lanes)
        v = fmaxf(v, __shfl_xor(v, 2, 32));
        if ((lane & 3) == 0) act1[outbase[u] + oc * 49] = (_Float16)v;
      }
    }
  }
}

// ---- fc1 GEMM via WMMA: (4096 x 3136) @ (3136 x 128) + bias, relu ---------
// block = one M-tile (16 batch rows, 100KB staged in LDS, shared by 8 waves),
// wave = one N-tile of 16 neurons streaming its weight columns from global.
__global__ void __launch_bounds__(256)
k_fc1_wmma(const _Float16* __restrict__ act1, const _Float16* __restrict__ w16,
           const float* __restrict__ bias, _Float16* __restrict__ out) {
  __shared__ _Float16 sAct[16 * 3136];  // 100352 B

  int mtile = blockIdx.x;  // 0..255
  {
    unsigned lds0 = (unsigned)(unsigned long long)&sAct[0];
    unsigned long long g0 = (unsigned long long)(act1 + mtile * 16 * 3136);
    for (unsigned chunk = threadIdx.x; chunk < 6272u; chunk += 256u)
      async_copy_b128(lds0 + chunk * 16u, g0 + (unsigned long long)chunk * 16ull);
    wait_async0();
  }
  __syncthreads();

  int lane = threadIdx.x & 31;
  int ntile = threadIdx.x >> 5;  // 0..7
  int part = lane >> 4;
  int ncol = ntile * 16 + (lane & 15);

  const _Float16* arow = &sAct[(lane & 15) * 3136];
  const _Float16* brow = w16 + ncol * 3136 + part * 16;

  v8f acc = {0.f, 0.f, 0.f, 0.f, 0.f, 0.f, 0.f, 0.f};
  for (int kb = 0; kb < 3136; kb += 32) {
    // A layout: halfs h -> K = kb + part*16 + h
    v8h lo = *(const v8h*)(arow + kb + part * 16);
    v8h hi = *(const v8h*)(arow + kb + part * 16 + 8);
    v16h a = __builtin_shufflevector(lo, hi, 0, 1, 2, 3, 4, 5, 6, 7,
                                     8, 9, 10, 11, 12, 13, 14, 15);
    // B layout: halfs h -> K = kb + part*16 + h, N = lane&15
    v16h bm = *(const v16h*)(brow + kb);
    acc = __builtin_amdgcn_wmma_f32_16x16x32_f16(
        false, a, false, bm, (short)0, acc, false, false);
  }
#pragma unroll
  for (int r = 0; r < 8; ++r) {
    int m = mtile * 16 + r + part * 8;
    int nn = ntile * 16 + (lane & 15);
    float v = fmaxf(acc[r] + bias[nn], 0.0f);
    out[m * 128 + nn] = (_Float16)v;
  }
}

// ---- fc2: 128 -> 10, fp32 out ---------------------------------------------
__global__ void k_fc2(const _Float16* __restrict__ a, const float* __restrict__ w,
                      const float* __restrict__ bias, float* __restrict__ out) {
  int t = blockIdx.x * blockDim.x + threadIdx.x;
  if (t >= BATCH * 10) return;
  int b = t / 10, o = t % 10;
  const _Float16* ar = a + b * 128;
  const float* wr = w + o * 128;
  float s = bias[o];
#pragma unroll 8
  for (int k = 0; k < 128; ++k) s = fmaf((float)ar[k], wr[k], s);
  out[t] = s;
}

extern "C" void kernel_launch(void* const* d_in, const int* in_sizes, int n_in,
                              void* d_out, int out_size, void* d_ws, size_t ws_size,
                              hipStream_t stream) {
  const float* x      = (const float*)d_in[0];
  const float* theta  = (const float*)d_in[1];
  const float* sigma  = (const float*)d_in[2];
  const float* gamma  = (const float*)d_in[3];
  const float* lambd  = (const float*)d_in[4];
  const float* psi    = (const float*)d_in[5];
  const float* conv2w = (const float*)d_in[6];
  const float* conv2b = (const float*)d_in[7];
  const float* fc1w   = (const float*)d_in[8];
  const float* fc1b   = (const float*)d_in[9];
  const float* fc2w   = (const float*)d_in[10];
  const float* fc2b   = (const float*)d_in[11];

  char* ws = (char*)d_ws;
  float*    w1     = (float*)(ws + 0);
  _Float16* afrag  = (_Float16*)(ws + 8192);
  _Float16* fc1w16 = (_Float16*)(ws + 65536);
  _Float16* xpad   = (_Float16*)(ws + 1048576u);
  _Float16* act1   = (_Float16*)(ws + 1048576u + 67108864u);
  _Float16* fc1o   = (_Float16*)(ws + 1048576u + 67108864u + 25690112u);

  k_gabor<<<1, 288, 0, stream>>>(theta, sigma, gamma, lambd, psi, w1);
  k_repack_conv2w<<<72, 256, 0, stream>>>(conv2w, afrag);
  k_f32_to_f16<<<(128 * 3136 + 255) / 256, 256, 0, stream>>>(fc1w, fc1w16, 128 * 3136);
  k_zero<<<16384, 256, 0, stream>>>((uint4*)xpad, 4194304);          // 64 MiB
  k_conv1<<<3136, 256, 0, stream>>>(x, w1, xpad);                    // 802816 threads
  k_conv2_wmma<<<3136, 256, 0, stream>>>(xpad, afrag, conv2b, act1); // 25088 waves x 2 tiles
  k_fc1_wmma<<<256, 256, 0, stream>>>(act1, fc1w16, fc1b, fc1o);     // 256 blocks x 8 waves
  k_fc2<<<160, 256, 0, stream>>>(fc1o, fc2w, fc2b, (float*)d_out);
}
